// Attention_6897717477568
// MI455X (gfx1250) — matrix operand order repaired and checked by reading, not compiled, not run on hardware
//
#include <hip/hip_runtime.h>
#include <hip/hip_bf16.h>
#include <math.h>

// ---- problem dims ----
#define B_    64
#define QL_   40
#define AL_   60
#define D_    300
#define C_    256
#define H_    256
#define KW_   3
#define HLU_  512
#define AH_   256
#define NC_   2
#define KCONV    900      // KW_*D_
#define KCONV_P  928      // padded to 29*32
#define KQA      556      // H_+D_
#define KQA_P    576      // 18*32
#define K1       768      // 3*H_, 24*32
#define NROWS_Q (B_*QL_)  // 2560
#define NROWS_A (B_*AL_)  // 3840

typedef _Float16 h16v __attribute__((ext_vector_type(16)));
typedef _Float16 h8v  __attribute__((ext_vector_type(8)));
typedef float    f8v  __attribute__((ext_vector_type(8)));

// ---------------- WMMA fragment loaders ----------------
// A (16x32 f16): lanes 0-15 hold row M=lane, K {0..7,16..23}; lanes 16-31 row M=lane-16, K {8..15,24..31}
__device__ __forceinline__ h16v frag_a_lds(const _Float16* base, int k, int lane) {
  int r = lane & 15, kb = (lane >> 4) << 3;
  const _Float16* p = base + r * D_ + k + kb;
  h16v o;
#pragma unroll
  for (int i = 0; i < 8; ++i) { o[i] = p[i]; o[8 + i] = p[16 + i]; }
  return o;
}
__device__ __forceinline__ h16v frag_a_g(const _Float16* base, int ld, int k, int lane) {
  int r = lane & 15, kb = (lane >> 4) << 3;
  const _Float16* p = base + (size_t)r * ld + k + kb;
  h8v lo = *(const h8v*)p;          // 16B aligned by construction
  h8v hi = *(const h8v*)(p + 16);
  h16v o;
#pragma unroll
  for (int i = 0; i < 8; ++i) { o[i] = lo[i]; o[8 + i] = hi[i]; }
  return o;
}
// B (32x16 f16) from K-major (transposed) weights: lane col = lane&15, K {0..7,16..23} lanes 0-15, {8..15,24..31} lanes 16-31.
__device__ __forceinline__ h16v frag_b(const _Float16* colmaj, int kstride, int col, int k, int lane) {
  const _Float16* p = colmaj + (size_t)col * kstride + k + ((lane >> 4) << 3);
  h8v lo = *(const h8v*)p;          // 16B aligned
  h8v hi = *(const h8v*)(p + 16);
  h16v o;
#pragma unroll
  for (int i = 0; i < 8; ++i) { o[i] = lo[i]; o[8 + i] = hi[i]; }
  return o;
}

// ---------------- prep: weight transpose + pad + f16 ----------------
// src: [K][N] f32 row-major  ->  dst: [N][Kpad] f16 (zero-padded K)
__global__ void k_prep_wt(const float* __restrict__ src, _Float16* __restrict__ dst,
                          int K, int N, int Kpad) {
  int idx = blockIdx.x * blockDim.x + threadIdx.x;
  int total = N * Kpad;
  if (idx >= total) return;
  int n = idx / Kpad, kp = idx - n * Kpad;
  dst[idx] = (_Float16)(kp < K ? src[(size_t)kp * N + n] : 0.f);
}

// ---------------- embedding gather (f16) + row norms ----------------
__global__ void k_embed(const int* __restrict__ question, const int* __restrict__ answer,
                        const float* __restrict__ emb,
                        _Float16* __restrict__ s1h, _Float16* __restrict__ s2h,
                        float* __restrict__ nq, float* __restrict__ na) {
  int row = blockIdx.x, tid = threadIdx.x;   // 64 threads
  bool isQ = row < NROWS_Q;
  int r = isQ ? row : row - NROWS_Q;
  int tok = isQ ? question[r] : answer[r];
  const float* e = emb + (size_t)tok * D_;
  _Float16* dst = (isQ ? s1h : s2h) + (size_t)r * D_;
  float ss = 0.f;
  for (int d = tid; d < D_; d += 64) { float v = e[d]; dst[d] = (_Float16)v; ss += v * v; }
  __shared__ float red[64];
  red[tid] = ss; __syncthreads();
  for (int s = 32; s > 0; s >>= 1) { if (tid < s) red[tid] += red[tid + s]; __syncthreads(); }
  if (tid == 0) { float n = sqrtf(red[0]); if (isQ) nq[r] = n; else na[r] = n; }
}

// ---------------- cosine matrix ----------------
__global__ void k_cos(const _Float16* __restrict__ s1h, const _Float16* __restrict__ s2h,
                      const float* __restrict__ nq, const float* __restrict__ na,
                      float* __restrict__ cosm) {
  int idx = blockIdx.x * blockDim.x + threadIdx.x;
  if (idx >= B_ * QL_ * AL_) return;
  int a = idx % AL_, bq = idx / AL_, b = bq / QL_;
  const _Float16* x = s1h + (size_t)bq * D_;
  const _Float16* y = s2h + (size_t)(b * AL_ + a) * D_;
  float dot = 0.f;
  for (int d = 0; d < D_; ++d) dot += (float)x[d] * (float)y[d];
  float den = fmaxf(nq[bq] * na[b * AL_ + a], 1e-6f);
  cosm[idx] = dot / den;
}

// ---------------- conv + relu + time-maxpool core (WMMA implicit GEMM) ----------------
// ks-outer / mt-inner: one B fragment feeds 4 M-tile accumulators (4x less L2 weight traffic).
#define ABUF_E (64 * D_ + 960)   // rows padded to 64, +K overread pad (zero-filled)

__device__ void conv_pool_wmma(const _Float16* Abuf, const _Float16* __restrict__ Wt,
                               const float* __restrict__ convb, int outRows,
                               _Float16* __restrict__ pooledOut) {
  int lane = threadIdx.x & 31, wave = threadIdx.x >> 5;
  for (int nt = wave; nt < C_ / 16; nt += 4) {
    int col = nt * 16 + (lane & 15);
    f8v acc[4];
#pragma unroll
    for (int mt = 0; mt < 4; ++mt) { f8v z = {}; acc[mt] = z; }
    for (int ks = 0; ks < KCONV_P / 32; ++ks) {
      h16v b = frag_b(Wt, KCONV_P, col, ks * 32, lane);
#pragma unroll
      for (int mt = 0; mt < 4; ++mt) {
        h16v a = frag_a_lds(Abuf + mt * 16 * D_, ks * 32, lane);
        acc[mt] = __builtin_amdgcn_wmma_f32_16x16x32_f16(false, a, false, b, (short)0, acc[mt], false, false);
      }
    }
    float bias = convb[col];
    float pmax = 0.f;                       // relu output >= 0
#pragma unroll
    for (int mt = 0; mt < 4; ++mt) {
      int rbase = mt * 16 + ((lane >> 4) << 3);
#pragma unroll
      for (int r = 0; r < 8; ++r)
        if (rbase + r < outRows) pmax = fmaxf(pmax, fmaxf(acc[mt][r] + bias, 0.f));
    }
    pmax = fmaxf(pmax, __shfl_xor(pmax, 16, 32));   // merge row halves (same col)
    if (lane < 16) pooledOut[col] = (_Float16)pmax;
  }
}

// context CNN over attention-gated answer, one WG per (b,q); gated tile built in LDS
__global__ void __launch_bounds__(128)
k_conv_ctx(const _Float16* __restrict__ s2h, const float* __restrict__ cosm,
           const _Float16* __restrict__ Wt, const float* __restrict__ convb,
           _Float16* __restrict__ pooledCtx) {
  __shared__ _Float16 Abuf[ABUF_E];
  int bq = blockIdx.x, b = bq / QL_;
  const _Float16* s2 = s2h + (size_t)b * AL_ * D_;
  const float* crow = cosm + (size_t)bq * AL_;
  for (int i = threadIdx.x; i < ABUF_E; i += 128) {
    float v = 0.f;
    if (i < AL_ * D_) v = (float)s2[i] * crow[i / D_];
    Abuf[i] = (_Float16)v;
  }
  __syncthreads();
  conv_pool_wmma(Abuf, Wt, convb, AL_ - KW_ + 1, pooledCtx + (size_t)bq * C_);
}

// plain Kim-CNN, one WG per sentence
__global__ void __launch_bounds__(128)
k_conv_sm(const _Float16* __restrict__ sh, int L, const _Float16* __restrict__ Wt,
          const float* __restrict__ convb, _Float16* __restrict__ pooled) {
  __shared__ _Float16 Abuf[ABUF_E];
  int b = blockIdx.x, n = L * D_;
  const _Float16* s = sh + (size_t)b * n;
  for (int i = threadIdx.x; i < ABUF_E; i += 128)
    Abuf[i] = (i < n) ? s[i] : (_Float16)0.f;
  __syncthreads();
  conv_pool_wmma(Abuf, Wt, convb, L - KW_ + 1, pooled + (size_t)b * C_);
}

// ---------------- WMMA GEMM (N=256): out_h[M,256] = A[M,Kpad] @ Bt[256,Kpad]^T + bias ----------------
// ks-outer: one A fragment feeds the wave's 4 N-tile accumulators.
__global__ void __launch_bounds__(128)
k_gemm(const _Float16* __restrict__ A, int lda, const _Float16* __restrict__ Bt, int Kpad,
       const float* __restrict__ bias, _Float16* __restrict__ outH, int ldh) {
  int lane = threadIdx.x & 31, wave = threadIdx.x >> 5;
  int m0 = blockIdx.x * 16;
  f8v acc[4];
#pragma unroll
  for (int i = 0; i < 4; ++i) { f8v z = {}; acc[i] = z; }
  for (int ks = 0; ks < Kpad / 32; ++ks) {
    h16v a = frag_a_g(A + (size_t)m0 * lda, lda, ks * 32, lane);
#pragma unroll
    for (int i = 0; i < 4; ++i) {
      int col = (wave + 4 * i) * 16 + (lane & 15);
      h16v b = frag_b(Bt, Kpad, col, ks * 32, lane);
      acc[i] = __builtin_amdgcn_wmma_f32_16x16x32_f16(false, a, false, b, (short)0, acc[i], false, false);
    }
  }
  int rbase = (lane >> 4) << 3;
#pragma unroll
  for (int i = 0; i < 4; ++i) {
    int col = (wave + 4 * i) * 16 + (lane & 15);
    float bv = bias[col];
#pragma unroll
    for (int r = 0; r < 8; ++r)
      outH[(size_t)(m0 + rbase + r) * ldh + col] = (_Float16)(acc[i][r] + bv);
  }
}

// ---------------- qa concat tail: qa[row][256:576] = [sent1 row | zeros] ----------------
__global__ void k_pack_qa(const _Float16* __restrict__ s1h, _Float16* __restrict__ qa) {
  int idx = blockIdx.x * blockDim.x + threadIdx.x;
  int tail = KQA_P - C_;                 // 320
  if (idx >= NROWS_Q * tail) return;
  int row = idx / tail, c = idx - row * tail;
  qa[(size_t)row * KQA_P + C_ + c] = (c < D_) ? s1h[(size_t)row * D_ + c] : (_Float16)0.f;
}

// ---------------- attention MLP: score = tanh(qa@attnW + b) @ probW ----------------
__global__ void __launch_bounds__(128)
k_attn(const _Float16* __restrict__ qa, const _Float16* __restrict__ attnWt,
       const float* __restrict__ attnb, const float* __restrict__ probW,
       float* __restrict__ score) {
  __shared__ float ssum[4][16];
  int lane = threadIdx.x & 31, wave = threadIdx.x >> 5;
  int m0 = blockIdx.x * 16;
  f8v acc[4];
#pragma unroll
  for (int i = 0; i < 4; ++i) { f8v z = {}; acc[i] = z; }
  for (int ks = 0; ks < KQA_P / 32; ++ks) {
    h16v a = frag_a_g(qa + (size_t)m0 * KQA_P, KQA_P, ks * 32, lane);
#pragma unroll
    for (int i = 0; i < 4; ++i) {
      int col = (wave + 4 * i) * 16 + (lane & 15);
      h16v b = frag_b(attnWt, KQA_P, col, ks * 32, lane);
      acc[i] = __builtin_amdgcn_wmma_f32_16x16x32_f16(false, a, false, b, (short)0, acc[i], false, false);
    }
  }
  float s[8];
#pragma unroll
  for (int r = 0; r < 8; ++r) s[r] = 0.f;
#pragma unroll
  for (int i = 0; i < 4; ++i) {
    int col = (wave + 4 * i) * 16 + (lane & 15);
    float ab = attnb[col], pw = probW[col];
#pragma unroll
    for (int r = 0; r < 8; ++r) s[r] += tanhf(acc[i][r] + ab) * pw;
  }
#pragma unroll
  for (int r = 0; r < 8; ++r) {   // reduce across 16 lanes sharing a row set
    float v = s[r];
    v += __shfl_xor(v, 1, 32); v += __shfl_xor(v, 2, 32);
    v += __shfl_xor(v, 4, 32); v += __shfl_xor(v, 8, 32);
    s[r] = v;
  }
  if ((lane & 15) == 0) {
    int rb = (lane >> 4) << 3;
#pragma unroll
    for (int r = 0; r < 8; ++r) ssum[wave][rb + r] = s[r];
  }
  __syncthreads();
  if (threadIdx.x < 16)
    score[m0 + threadIdx.x] =
        ssum[0][threadIdx.x] + ssum[1][threadIdx.x] + ssum[2][threadIdx.x] + ssum[3][threadIdx.x];
}

// ---------------- softmax over QL + weighted feature3 ----------------
__global__ void k_softmax_f3(const float* __restrict__ score, const _Float16* __restrict__ qa,
                             _Float16* __restrict__ fcombH) {
  __shared__ float prob[QL_];
  int b = blockIdx.x, tid = threadIdx.x;   // 128 threads
  if (tid == 0) {
    float mx = -1e30f;
    for (int q = 0; q < QL_; ++q) mx = fmaxf(mx, score[b * QL_ + q]);
    float sum = 0.f;
    for (int q = 0; q < QL_; ++q) { float e = expf(score[b * QL_ + q] - mx); prob[q] = e; sum += e; }
    for (int q = 0; q < QL_; ++q) prob[q] /= sum;
  }
  __syncthreads();
  for (int h = tid; h < H_; h += 128) {
    float f3 = 0.f;
    for (int q = 0; q < QL_; ++q)
      f3 += prob[q] * (float)qa[(size_t)(b * QL_ + q) * KQA_P + h];   // feat lives in qa[:, :256]
    fcombH[(size_t)b * K1 + 2 * H_ + h] = (_Float16)f3;
  }
}

// ---------------- Linear(768,512) + BatchNorm(train) + tanh, single WG ----------------
__global__ void __launch_bounds__(128)
k_final(const _Float16* __restrict__ fcombH, const _Float16* __restrict__ W1t,
        const float* __restrict__ b1, const float* __restrict__ gamma,
        const float* __restrict__ beta, float* __restrict__ feat_out) {
  int lane = threadIdx.x & 31, wave = threadIdx.x >> 5;
  for (int nt = wave; nt < HLU_ / 16; nt += 4) {
    int col = nt * 16 + (lane & 15);
    f8v acc[4];
#pragma unroll
    for (int mt = 0; mt < 4; ++mt) { f8v z = {}; acc[mt] = z; }
    for (int ks = 0; ks < K1 / 32; ++ks) {
      h16v b = frag_b(W1t, K1, col, ks * 32, lane);
#pragma unroll
      for (int mt = 0; mt < 4; ++mt) {
        h16v a = frag_a_g(fcombH + (size_t)(mt * 16) * K1, K1, ks * 32, lane);
        acc[mt] = __builtin_amdgcn_wmma_f32_16x16x32_f16(false, a, false, b, (short)0, acc[mt], false, false);
      }
    }
    float bv = b1[col], cs = 0.f, cq = 0.f;
#pragma unroll
    for (int mt = 0; mt < 4; ++mt)
#pragma unroll
      for (int r = 0; r < 8; ++r) { float v = acc[mt][r] + bv; cs += v; cq += v * v; }
    cs += __shfl_xor(cs, 16, 32);     // other 32 rows live in partner lane
    cq += __shfl_xor(cq, 16, 32);
    float mean = cs * (1.f / B_);
    float var = cq * (1.f / B_) - mean * mean;      // biased (BN train mode)
    float scale = gamma[col] * rsqrtf(var + 1e-5f);
    float shift = beta[col] - mean * scale;
    int rb = (lane >> 4) << 3;
#pragma unroll
    for (int mt = 0; mt < 4; ++mt)
#pragma unroll
      for (int r = 0; r < 8; ++r)
        feat_out[(size_t)(mt * 16 + rb + r) * HLU_ + col] =
            tanhf((acc[mt][r] + bv) * scale + shift);
  }
}

// ---------------- head: preds = log_softmax(feat_out @ W2 + b2) ----------------
__global__ void k_head(const float* __restrict__ feat_out, const float* __restrict__ W2,
                       const float* __restrict__ b2, float* __restrict__ preds) {
  int m = threadIdx.x;
  if (m >= B_) return;
  float l0 = b2[0], l1 = b2[1];
  const float* x = feat_out + (size_t)m * HLU_;
  for (int k = 0; k < HLU_; ++k) { float v = x[k]; l0 += v * W2[2 * k]; l1 += v * W2[2 * k + 1]; }
  float mx = fmaxf(l0, l1);
  float lse = mx + logf(expf(l0 - mx) + expf(l1 - mx));
  preds[2 * m] = l0 - lse;
  preds[2 * m + 1] = l1 - lse;
}

// ---------------- launch ----------------
extern "C" void kernel_launch(void* const* d_in, const int* in_sizes, int n_in,
                              void* d_out, int out_size, void* d_ws, size_t ws_size,
                              hipStream_t stream) {
  const int*   question = (const int*)d_in[0];
  const int*   answer   = (const int*)d_in[1];
  const float* emb      = (const float*)d_in[3];   // d_in[2] ext_feats unused by reference
  const float* sm_convW = (const float*)d_in[4];
  const float* sm_convb = (const float*)d_in[5];
  const float* sm_fcW   = (const float*)d_in[6];
  const float* sm_fcb   = (const float*)d_in[7];
  const float* ctx_convW= (const float*)d_in[8];
  const float* ctx_convb= (const float*)d_in[9];
  const float* ctx_fcW  = (const float*)d_in[10];
  const float* ctx_fcb  = (const float*)d_in[11];
  const float* attnW    = (const float*)d_in[12];
  const float* attnb    = (const float*)d_in[13];
  const float* probW    = (const float*)d_in[14];
  const float* W1       = (const float*)d_in[15];
  const float* b1       = (const float*)d_in[16];
  const float* gamma_   = (const float*)d_in[17];
  const float* beta_    = (const float*)d_in[18];
  const float* W2       = (const float*)d_in[19];
  const float* b2       = (const float*)d_in[20];
  float* preds    = (float*)d_out;
  float* feat_out = (float*)d_out + B_ * NC_;

  char* ws = (char*)d_ws;
  size_t off = 0;
  auto alloc = [&](size_t bytes) -> char* { char* p = ws + off; off += (bytes + 255) & ~(size_t)255; return p; };
  _Float16* s1h       = (_Float16*)alloc((size_t)NROWS_Q * D_ * 2);
  _Float16* s2h       = (_Float16*)alloc((size_t)NROWS_A * D_ * 2);
  float*    nq        = (float*)   alloc((size_t)NROWS_Q * 4);
  float*    na        = (float*)   alloc((size_t)NROWS_A * 4);
  float*    cosm      = (float*)   alloc((size_t)NROWS_Q * AL_ * 4);
  _Float16* ctxWt     = (_Float16*)alloc((size_t)C_ * KCONV_P * 2);
  _Float16* smWt      = (_Float16*)alloc((size_t)C_ * KCONV_P * 2);
  _Float16* ctx_fcWt  = (_Float16*)alloc((size_t)H_ * C_ * 2);
  _Float16* sm_fcWt   = (_Float16*)alloc((size_t)H_ * C_ * 2);
  _Float16* attnWt    = (_Float16*)alloc((size_t)AH_ * KQA_P * 2);
  _Float16* W1t       = (_Float16*)alloc((size_t)HLU_ * K1 * 2);
  _Float16* pooled_ctx= (_Float16*)alloc((size_t)NROWS_Q * C_ * 2);
  _Float16* pooled_q  = (_Float16*)alloc((size_t)B_ * C_ * 2);
  _Float16* pooled_a  = (_Float16*)alloc((size_t)B_ * C_ * 2);
  _Float16* qa        = (_Float16*)alloc((size_t)NROWS_Q * KQA_P * 2);
  float*    score     = (float*)   alloc((size_t)NROWS_Q * 4);
  _Float16* fcombH    = (_Float16*)alloc((size_t)B_ * K1 * 2);

  auto blocks = [](int total, int bs) { return (total + bs - 1) / bs; };

  // weight prep (f16, K-major, padded)
  k_prep_wt<<<blocks(C_ * KCONV_P, 256), 256, 0, stream>>>(ctx_convW, ctxWt, KCONV, C_, KCONV_P);
  k_prep_wt<<<blocks(C_ * KCONV_P, 256), 256, 0, stream>>>(sm_convW, smWt, KCONV, C_, KCONV_P);
  k_prep_wt<<<blocks(H_ * C_, 256), 256, 0, stream>>>(ctx_fcW, ctx_fcWt, C_, H_, C_);
  k_prep_wt<<<blocks(H_ * C_, 256), 256, 0, stream>>>(sm_fcW, sm_fcWt, C_, H_, C_);
  k_prep_wt<<<blocks(AH_ * KQA_P, 256), 256, 0, stream>>>(attnW, attnWt, KQA, AH_, KQA_P);
  k_prep_wt<<<blocks(HLU_ * K1, 256), 256, 0, stream>>>(W1, W1t, K1, HLU_, K1);

  // embeddings + norms + cosine
  k_embed<<<NROWS_Q + NROWS_A, 64, 0, stream>>>(question, answer, emb, s1h, s2h, nq, na);
  k_cos<<<blocks(B_ * QL_ * AL_, 256), 256, 0, stream>>>(s1h, s2h, nq, na, cosm);

  // CNNs (WMMA implicit GEMM)
  k_conv_sm<<<B_, 128, 0, stream>>>(s1h, QL_, smWt, sm_convb, pooled_q);
  k_conv_sm<<<B_, 128, 0, stream>>>(s2h, AL_, smWt, sm_convb, pooled_a);
  k_conv_ctx<<<NROWS_Q, 128, 0, stream>>>(s2h, cosm, ctxWt, ctx_convb, pooled_ctx);

  // FC layers (WMMA GEMM) -> f16 feature buffers
  k_gemm<<<B_ / 16, 128, 0, stream>>>(pooled_q, C_, sm_fcWt, C_, sm_fcb, fcombH, K1);          // feature1
  k_gemm<<<B_ / 16, 128, 0, stream>>>(pooled_a, C_, sm_fcWt, C_, sm_fcb, fcombH + H_, K1);     // feature2
  k_gemm<<<NROWS_Q / 16, 128, 0, stream>>>(pooled_ctx, C_, ctx_fcWt, C_, ctx_fcb, qa, KQA_P);  // feat -> qa[:, :256]

  // qa tail, attention score, softmax + feature3
  k_pack_qa<<<blocks(NROWS_Q * (KQA_P - C_), 256), 256, 0, stream>>>(s1h, qa);
  k_attn<<<NROWS_Q / 16, 128, 0, stream>>>(qa, attnWt, attnb, probW, score);
  k_softmax_f3<<<B_, 128, 0, stream>>>(score, qa, fcombH);

  // final layers + head
  k_final<<<1, 128, 0, stream>>>(fcombH, W1t, b1, gamma_, beta_, feat_out);
  k_head<<<1, 64, 0, stream>>>(feat_out, W2, b2, preds);
}
